// Attention_4964982194702
// MI455X (gfx1250) — compile-verified, matrix operand order checked by reading, not run on hardware
//
#include <hip/hip_runtime.h>
#include <hip/hip_bf16.h>
#include <math.h>

#define H_   8
#define Q_   4
#define N_   1024
#define DIM_ 512
#define DH_  64
#define HQ_  (H_*Q_)
#define TOK_ (Q_*N_)          // 4096 tokens
#define MCNT 33554432.0       // 8*4*1024*1024 elements per score tensor
#define COVSCALE ((0.001f/1024.0f)/(8.0f+1e-4f))

typedef __attribute__((ext_vector_type(16))) __bf16 v16bf;
typedef __attribute__((ext_vector_type(8)))  __bf16 v8bf;
typedef __attribute__((ext_vector_type(8)))  float  v8f;

static __device__ __forceinline__ v16bf ldfrag(const __bf16* p, int lh) {
  // Per-lane A/B fragment: elements 0..7 <- p[8*lh .. 8*lh+7], 8..15 <- p[16+8*lh .. 23+8*lh]
  const __bf16* q = p + 8 * lh;
  v8bf lo = *(const v8bf*)(q);
  v8bf hi = *(const v8bf*)(q + 16);
  return __builtin_shufflevector(lo, hi, 0,1,2,3,4,5,6,7,8,9,10,11,12,13,14,15);
}
static __device__ __forceinline__ v8f wmma_bf16(v16bf a, v16bf b, v8f c) {
  return __builtin_amdgcn_wmma_f32_16x16x32_bf16(false, a, false, b, (short)0, c, false, false);
}
static __device__ __forceinline__ float clampf(float x, float lo, float hi) { return fminf(fmaxf(x, lo), hi); }
static __device__ __forceinline__ float redAdd32(float v) { for (int o = 1; o < 32; o <<= 1) v += __shfl_xor(v, o, 32); return v; }
static __device__ __forceinline__ float redAdd16(float v) { for (int o = 1; o < 16; o <<= 1) v += __shfl_xor(v, o, 32); return v; }
static __device__ __forceinline__ float redMax32(float v) { for (int o = 1; o < 32; o <<= 1) v = fmaxf(v, __shfl_xor(v, o, 32)); return v; }

#define ZERO8 {0.f,0.f,0.f,0.f,0.f,0.f,0.f,0.f}

// ---------------------------------------------------------------------------
// K_init: zero the double accumulators
// ---------------------------------------------------------------------------
__global__ void zero_accum_kernel(double* a) { if (threadIdx.x < 6) a[threadIdx.x] = 0.0; }

// ---------------------------------------------------------------------------
// K0: transpose + convert weights to bf16 (B-fragment friendly: [out_col][k])
// ---------------------------------------------------------------------------
__global__ void convw_kernel(const float* __restrict__ W_in, const float* __restrict__ W_out,
                             __bf16* __restrict__ WinT, __bf16* __restrict__ WoutT) {
  int idx = blockIdx.x * blockDim.x + threadIdx.x;
  if (idx < 262144) {
    int c = idx >> 9, k = idx & 511;
    WinT[c * 512 + k] = (__bf16)W_in[k * 512 + c];
  } else if (idx < 524288) {
    int j = idx - 262144;
    int c = j >> 9, k = j & 511;
    WoutT[c * 512 + k] = (__bf16)W_out[k * 512 + c];
  }
}

// ---------------------------------------------------------------------------
// K1: fused LayerNorm + input projection (bf16 WMMA).
//   mode 0/1: fdst = [h][q][n][d] row-major + per-row norm/sum stats
//   mode 2  : fdst = [h][q][d][n] (transposed, for PV B-fragments), no stats
// Block: 16 tokens; 8 waves, wave w = head w (cols w*64..w*64+63).
// ---------------------------------------------------------------------------
__global__ __launch_bounds__(256)
void proj_kernel(const float* __restrict__ x, const float* __restrict__ ln_w, const float* __restrict__ ln_b,
                 const __bf16* __restrict__ WinT, __bf16* __restrict__ fdst,
                 float* __restrict__ norm_out, float* __restrict__ sum_out, int mode) {
  __shared__ float  xs[16 * 512];   // 32 KB
  __shared__ __bf16 xn[16 * 512];   // 16 KB
  const int qb = blockIdx.y, n0 = blockIdx.x * 16;
  const int tid = threadIdx.x, wave = tid >> 5, lane = tid & 31;
  const int lr = lane & 15, lh = lane >> 4;

  // Warm the weight columns this wave will stream (gfx1250 global_prefetch_b8)
  __builtin_prefetch(WinT + (size_t)(wave * 64 + lr) * 512, 0, 1);

  for (int idx = tid; idx < 16 * 512; idx += 256)
    xs[idx] = x[((size_t)qb * N_ + n0 + (idx >> 9)) * DIM_ + (idx & 511)];
  __syncthreads();

  // LayerNorm: wave w handles rows 2w, 2w+1
  for (int rr = 0; rr < 2; rr++) {
    int r = wave * 2 + rr;
    float s = 0.f, s2 = 0.f;
    for (int i = lane; i < 512; i += 32) { float v = xs[r * 512 + i]; s += v; s2 += v * v; }
    s = redAdd32(s); s2 = redAdd32(s2);
    float mean = s * (1.f / 512.f);
    float var  = s2 * (1.f / 512.f) - mean * mean;
    float inv  = rsqrtf(var + 1e-5f);
    for (int i = lane; i < 512; i += 32) {
      float v = (xs[r * 512 + i] - mean) * inv * ln_w[i] + ln_b[i];
      xn[r * 512 + i] = (__bf16)v;
    }
  }
  __syncthreads();

  const int hq = wave * Q_ + qb;     // head == wave
  float p1[8], p2[8];
#pragma unroll
  for (int j = 0; j < 8; j++) { p1[j] = 0.f; p2[j] = 0.f; }

  const __bf16* arow = xn + lr * 512;
  for (int ti = 0; ti < 4; ti++) {
    int c0 = wave * 64 + ti * 16;
    v8f acc = ZERO8;
    const __bf16* brow = WinT + (size_t)(c0 + lr) * 512;
    // software pipeline: prefetch next-k fragments before consuming current
    v16bf a = ldfrag(arow, lh);
    v16bf b = ldfrag(brow, lh);
    for (int kk = 0; kk < 16; kk++) {
      int kn = ((kk + 1) & 15) * 32;
      v16bf na = ldfrag(arow + kn, lh);
      v16bf nb = ldfrag(brow + kn, lh);
      acc = wmma_bf16(a, b, acc);
      a = na; b = nb;
    }
#pragma unroll
    for (int j = 0; j < 8; j++) {
      float v = acc[j];
      int r = j + 8 * lh;
      int n = n0 + r;
      int d = ti * 16 + lr;
      if (mode == 2) fdst[((size_t)hq * DH_ + d) * N_ + n] = (__bf16)v;
      else           fdst[((size_t)hq * N_ + n) * DH_ + d] = (__bf16)v;
      p1[j] += v; p2[j] += v * v;
    }
  }
  if (mode != 2) {
#pragma unroll
    for (int j = 0; j < 8; j++) {
      float s1 = redAdd16(p1[j]);
      float s2 = redAdd16(p2[j]);
      if (lr == 0) {
        int r = j + 8 * lh;
        norm_out[(size_t)hq * N_ + n0 + r] = sqrtf(s2);
        sum_out [(size_t)hq * N_ + n0 + r] = s1;
      }
    }
  }
}

// ---------------------------------------------------------------------------
// K2a: per-(h,q) key means over tokens. 2048 threads: (hq,d)
// ---------------------------------------------------------------------------
__global__ void colmean_kernel(const __bf16* __restrict__ fk, float* __restrict__ fkmean) {
  int idx = blockIdx.x * blockDim.x + threadIdx.x;
  if (idx >= HQ_ * DH_) return;
  int hq = idx >> 6, d = idx & 63;
  const __bf16* p = fk + (size_t)hq * N_ * DH_ + d;
  float s = 0.f;
  for (int n = 0; n < N_; n++) s += (float)p[(size_t)n * DH_];
  fkmean[idx] = s * (1.f / (float)N_);
}
__global__ void chq_kernel(const float* __restrict__ fkmean, float* __restrict__ chq) {
  int hq = threadIdx.x;
  if (hq < HQ_) { float s = 0.f; for (int d = 0; d < DH_; d++) s += fkmean[hq * DH_ + d]; chq[hq] = s; }
}

// K2b: per-head MLP feature means (over q,n). 1024 threads.
__global__ void feat_kernel(const __bf16* __restrict__ fq, const __bf16* __restrict__ fk, float* __restrict__ feat) {
  int idx = blockIdx.x * blockDim.x + threadIdx.x;
  if (idx >= 1024) return;
  int which = idx >> 9, hd = idx & 511, h = hd >> 6, d = hd & 63;
  const __bf16* src = which ? fk : fq;
  float s = 0.f;
  for (int qq = 0; qq < Q_; qq++) {
    const __bf16* p = src + ((size_t)(h * Q_ + qq) * N_) * DH_ + d;
    for (int n = 0; n < N_; n++) s += (float)p[(size_t)n * DH_];
  }
  feat[h * 128 + which * 64 + d] = s * (1.f / (float)TOK_);
}

// K2c: t[hq][n] = dot(fq_n, fkmean_hq). 32768 threads.
__global__ void tdot_kernel(const __bf16* __restrict__ fq, const float* __restrict__ fkmean, float* __restrict__ tdot) {
  int idx = blockIdx.x * blockDim.x + threadIdx.x;
  if (idx >= HQ_ * N_) return;
  int hq = idx >> 10;
  const __bf16* p = fq + (size_t)idx * DH_;
  const float* m = fkmean + hq * DH_;
  float s = 0.f;
  for (int d = 0; d < DH_; d++) s += (float)p[d] * m[d];
  tdot[idx] = s;
}

// ---------------------------------------------------------------------------
// K3: weight-predictor MLP (8 heads, one wave each)
// ---------------------------------------------------------------------------
__global__ void mlp_kernel(const float* __restrict__ feat,
                           const float* __restrict__ W1, const float* __restrict__ b1,
                           const float* __restrict__ lnw, const float* __restrict__ lnb,
                           const float* __restrict__ W2, const float* __restrict__ b2,
                           const float* __restrict__ W3, const float* __restrict__ b3,
                           const float* __restrict__ W4, const float* __restrict__ b4,
                           const float* __restrict__ wtemp, float* __restrict__ wout) {
  __shared__ float h1[8][256];
  __shared__ float h2[8][192];
  __shared__ float h3[8][64];
  const int wave = threadIdx.x >> 5, lane = threadIdx.x & 31;
  const int h = wave;
  const float* f = feat + h * 128;

  for (int j = lane; j < 256; j += 32) {
    float s = b1[j];
    for (int i = 0; i < 128; i++) s += f[i] * W1[i * 256 + j];
    h1[h][j] = s;
  }
  __syncthreads();
  float s = 0.f, s2 = 0.f;
  for (int j = lane; j < 256; j += 32) { float v = h1[h][j]; s += v; s2 += v * v; }
  s = redAdd32(s); s2 = redAdd32(s2);
  float mean = s * (1.f / 256.f), var = s2 * (1.f / 256.f) - mean * mean;
  float inv = rsqrtf(var + 1e-5f);
  for (int j = lane; j < 256; j += 32) {
    float v = (h1[h][j] - mean) * inv * lnw[j] + lnb[j];
    h1[h][j] = fmaxf(v, 0.f);
  }
  __syncthreads();
  for (int j = lane; j < 192; j += 32) {
    float t = b2[j];
    for (int i = 0; i < 256; i++) t += h1[h][i] * W2[i * 192 + j];
    h2[h][j] = fmaxf(t, 0.f);
  }
  __syncthreads();
  for (int j = lane; j < 64; j += 32) {
    float t = b3[j];
    for (int i = 0; i < 192; i++) t += h2[h][i] * W3[i * 64 + j];
    h3[h][j] = fmaxf(t, 0.f);
  }
  __syncthreads();
  if (lane == 0) {
    float lg[3];
    for (int c = 0; c < 3; c++) {
      float t = b4[c];
      for (int i = 0; i < 64; i++) t += h3[h][i] * W4[i * 3 + c];
      lg[c] = t;
    }
    float mx = fmaxf(lg[0], fmaxf(lg[1], lg[2]));
    float e0 = expf(lg[0] - mx), e1 = expf(lg[1] - mx), e2 = expf(lg[2] - mx);
    float den = e0 + e1 + e2;
    float p0 = e0 / den, p1 = e1 / den, p2 = e2 / den;
    float wt = clampf(wtemp[0], 0.01f, 1.0f);
    float q0 = p0 / wt, q1 = p1 / wt, q2 = p2 / wt;
    float m2 = fmaxf(q0, fmaxf(q1, q2));
    float f0 = expf(q0 - m2), f1 = expf(q1 - m2), f2 = expf(q2 - m2);
    float dn = f0 + f1 + f2;
    float w0 = clampf(f0 / dn, 0.01f, 0.95f);
    float w1 = clampf(f1 / dn, 0.01f, 0.95f);
    float w2 = clampf(f2 / dn, 0.01f, 0.95f);
    float ws = w0 + w1 + w2;
    wout[h * 3 + 0] = w0 / ws;
    wout[h * 3 + 1] = w1 / ws;
    wout[h * 3 + 2] = w2 / ws;
  }
}

// ---------------------------------------------------------------------------
// K4/K6: score strip pass.  Block = 16 query rows x 1024 keys, 8 waves.
//   mode 0: accumulate cos & cov sum/sumsq (accum[0..3]), write var_row.
//   mode 1: accumulate dots sum/sumsq (accum[4..5]).
// Pipelined: next tile's B fragments are issued before current WMMAs retire.
// ---------------------------------------------------------------------------
__global__ __launch_bounds__(256)
void score_pass_kernel(const __bf16* __restrict__ fq, const __bf16* __restrict__ fk,
                       const float* __restrict__ nq, const float* __restrict__ sq,
                       const float* __restrict__ nk, const float* __restrict__ sk,
                       const float* __restrict__ tdot, const float* __restrict__ chq,
                       float* __restrict__ var_row, const float* __restrict__ coefs,
                       double* __restrict__ accum, int mode) {
  __shared__ float lds_m[16];
  __shared__ float sred[8];
  const int strip = blockIdx.x, qb = blockIdx.y, h = blockIdx.z;
  const int hq = h * Q_ + qb;
  const int n0 = strip * 16;
  const int tid = threadIdx.x, wave = tid >> 5, lane = tid & 31;
  const int lr = lane & 15, lh = lane >> 4;

  if (tid < 16) lds_m[tid] = 0.f;
  __syncthreads();

  const __bf16* fqr = fq + ((size_t)hq * N_ + n0 + lr) * DH_;
  const __bf16* fkb = fk + (size_t)hq * N_ * DH_;

  // warm the wave's 128-key span (global_prefetch_b8), 512B apart per lane
  __builtin_prefetch(fkb + (size_t)(wave * 128 + lane * 4) * DH_, 0, 1);

  v16bf a0 = ldfrag(fqr + 0, lh);
  v16bf a1 = ldfrag(fqr + 32, lh);

  const int rowbase = hq * N_ + n0 + 8 * lh;
  float r_nq[8], r_mu[8], r_t[8], r_cc[8];
#pragma unroll
  for (int j = 0; j < 8; j++) {
    r_nq[j] = nq[rowbase + j];
    r_mu[j] = sq[rowbase + j] * (1.f / 64.f);
    r_t[j]  = tdot[rowbase + j];
    r_cc[j] = 0.f;
  }
  const float chqv = chq[hq];
  float ca = 0.f, cb = 0.f;
  if (mode == 1) {
    ca = coefs[h]; cb = coefs[8 + h];
    float cc = coefs[16 + h];
#pragma unroll
    for (int j = 0; j < 8; j++) r_cc[j] = cc * var_row[rowbase + j];
  }

  float s0 = 0.f, s1 = 0.f, s2 = 0.f, s3 = 0.f;
  float marg[8];
#pragma unroll
  for (int j = 0; j < 8; j++) marg[j] = 0.f;

  const __bf16* fkr0 = fkb + (size_t)(wave * 128 + lr) * DH_;
  v16bf b0 = ldfrag(fkr0 + 0, lh);
  v16bf b1 = ldfrag(fkr0 + 32, lh);

  for (int t = 0; t < 8; t++) {
    int m0 = wave * 128 + t * 16;
    // prefetch next tile's fragments (wrap keeps control flow uniform)
    const __bf16* fkrn = fkb + (size_t)(wave * 128 + ((t + 1) & 7) * 16 + lr) * DH_;
    v16bf nb0 = ldfrag(fkrn + 0, lh);
    v16bf nb1 = ldfrag(fkrn + 32, lh);
    v8f acc = ZERO8;
    acc = wmma_bf16(a0, b0, acc);
    acc = wmma_bf16(a1, b1, acc);
    int mcol = hq * N_ + m0 + lr;
    float c_nk = nk[mcol], c_sk = sk[mcol];
#pragma unroll
    for (int j = 0; j < 8; j++) {
      float S = acc[j];
      float cosv = clampf(S / ((r_nq[j] + 1e-6f) * (c_nk + 1e-6f)), -0.95f, 0.95f);
      float covv = clampf((S - r_t[j] - r_mu[j] * (c_sk - chqv)) * COVSCALE, -50.f, 50.f);
      if (mode == 0) {
        float csn = clampf(S / (fmaxf(r_nq[j], 1e-4f) * fmaxf(c_nk, 1e-4f)), -0.95f, 0.95f);
        float mg = clampf(0.01f - csn, 0.f, 15.f);
        s0 += cosv; s1 += cosv * cosv;
        s2 += covv; s3 += covv * covv;
        marg[j] += mg;
      } else {
        float d = ca * cosv + cb * covv + r_cc[j];
        s0 += d; s1 += d * d;
      }
    }
    b0 = nb0; b1 = nb1;
  }

  if (mode == 0) {
#pragma unroll
    for (int j = 0; j < 8; j++) {
      float ms = redAdd16(marg[j]);
      if (lr == 0) atomicAdd(&lds_m[j + 8 * lh], ms);
    }
    __syncthreads();
    if (tid < 16) var_row[hq * N_ + n0 + tid] = lds_m[tid] * (1.f / (float)N_);
  }

  float vals[4] = { s0, s1, s2, s3 };
  int nvals = (mode == 0) ? 4 : 2;
  int base = (mode == 0) ? 0 : 4;
  for (int i = 0; i < nvals; i++) {
    float v = redAdd32(vals[i]);
    if ((tid & 31) == 0) sred[tid >> 5] = v;
    __syncthreads();
    if (tid == 0) {
      float s = 0.f;
      for (int wv = 0; wv < 8; wv++) s += sred[wv];
      atomicAdd(&accum[base + i], (double)s);
    }
    __syncthreads();
  }
}

// ---------------------------------------------------------------------------
// K5: component stds + per-head combination coefficients
// ---------------------------------------------------------------------------
__global__ void finalize1_kernel(const float* __restrict__ var_row, const double* __restrict__ accum,
                                 const float* __restrict__ wht, float* __restrict__ coefs) {
  __shared__ float sred[8];
  __shared__ float gsum[1];
  float s = 0.f, s2 = 0.f;
  for (int i = threadIdx.x; i < HQ_ * N_; i += 256) { float v = var_row[i]; s += v; s2 += v * v; }
  float t1 = redAdd32(s);
  if ((threadIdx.x & 31) == 0) sred[threadIdx.x >> 5] = t1;
  __syncthreads();
  if (threadIdx.x == 0) { float t = 0.f; for (int i = 0; i < 8; i++) t += sred[i]; gsum[0] = t; }
  __syncthreads();
  float t2 = redAdd32(s2);
  if ((threadIdx.x & 31) == 0) sred[threadIdx.x >> 5] = t2;
  __syncthreads();
  if (threadIdx.x == 0) {
    float totS2 = 0.f;
    for (int i = 0; i < 8; i++) totS2 += sred[i];
    double M = MCNT;
    double vs = (double)gsum[0] * 1024.0, vss = (double)totS2 * 1024.0;  // broadcast over m
    double stdvar = sqrt(fmax(0.0, (vss - vs * vs / M) / (M - 1.0)));
    double cs = accum[0], css = accum[1], vv = accum[2], vvs = accum[3];
    double stdcos = sqrt(fmax(0.0, (css - cs * cs / M) / (M - 1.0)));
    double stdcov = sqrt(fmax(0.0, (vvs - vv * vv / M) / (M - 1.0)));
    for (int h = 0; h < 8; h++) {
      coefs[h]      = wht[h * 3 + 0] / ((float)stdcos + 1e-4f);
      coefs[8 + h]  = wht[h * 3 + 1] / ((float)stdcov + 1e-4f) * 0.3f;
      coefs[16 + h] = wht[h * 3 + 2] / ((float)stdvar + 1e-4f) * 0.3f;
    }
  }
}

// K7: temperature
__global__ void finalize2_kernel(const double* __restrict__ accum, float* __restrict__ invTemp) {
  if (threadIdx.x == 0) {
    double M = MCNT;
    double s = accum[4], ss = accum[5];
    double ds = sqrt(fmax(0.0, (ss - s * s / M) / (M - 1.0)));
    float temp = (ds < 1e-5) ? 0.01f : ((ds < 1e-3) ? 0.05f : (0.2f + (float)ds * 2.0f));
    temp = clampf(temp, 0.01f, 8.0f);
    invTemp[0] = 1.0f / temp;
  }
}

// ---------------------------------------------------------------------------
// K8: recompute dots strip -> softmax -> attn @ V (WMMA), write fhead bf16
// ---------------------------------------------------------------------------
__global__ __launch_bounds__(256)
void softmax_pv_kernel(const __bf16* __restrict__ fq, const __bf16* __restrict__ fk,
                       const __bf16* __restrict__ fvT,
                       const float* __restrict__ nq, const float* __restrict__ sq,
                       const float* __restrict__ nk, const float* __restrict__ sk,
                       const float* __restrict__ tdot, const float* __restrict__ chq,
                       const float* __restrict__ var_row, const float* __restrict__ coefs,
                       const float* __restrict__ invTemp, __bf16* __restrict__ fheadB) {
  __shared__ __bf16 dotsS[16 * 1024];   // 32 KB
  __shared__ float part[1024];          // 4 KB  (cross-wave K-split partials)
  __shared__ float rowMax[16], rowSI[16];
  const int strip = blockIdx.x, qb = blockIdx.y, h = blockIdx.z;
  const int hq = h * Q_ + qb;
  const int n0 = strip * 16;
  const int tid = threadIdx.x, wave = tid >> 5, lane = tid & 31;
  const int lr = lane & 15, lh = lane >> 4;
  const float invT = invTemp[0];

  const __bf16* fqr = fq + ((size_t)hq * N_ + n0 + lr) * DH_;
  const __bf16* fkb = fk + (size_t)hq * N_ * DH_;

  __builtin_prefetch(fkb + (size_t)(wave * 128 + lane * 4) * DH_, 0, 1);
  __builtin_prefetch(fvT + ((size_t)hq * DH_ + (wave & 3) * 16 + lr) * N_, 0, 1);

  v16bf a0 = ldfrag(fqr + 0, lh);
  v16bf a1 = ldfrag(fqr + 32, lh);

  const int rowbase = hq * N_ + n0 + 8 * lh;
  float r_nq[8], r_mu[8], r_t[8], r_cc[8];
  const float ca = coefs[h], cb = coefs[8 + h], cc = coefs[16 + h];
#pragma unroll
  for (int j = 0; j < 8; j++) {
    r_nq[j] = nq[rowbase + j];
    r_mu[j] = sq[rowbase + j] * (1.f / 64.f);
    r_t[j]  = tdot[rowbase + j];
    r_cc[j] = cc * var_row[rowbase + j];
  }
  const float chqv = chq[hq];

  const __bf16* fkr0 = fkb + (size_t)(wave * 128 + lr) * DH_;
  v16bf b0 = ldfrag(fkr0 + 0, lh);
  v16bf b1 = ldfrag(fkr0 + 32, lh);

  for (int t = 0; t < 8; t++) {
    int m0 = wave * 128 + t * 16;
    const __bf16* fkrn = fkb + (size_t)(wave * 128 + ((t + 1) & 7) * 16 + lr) * DH_;
    v16bf nb0 = ldfrag(fkrn + 0, lh);
    v16bf nb1 = ldfrag(fkrn + 32, lh);
    v8f acc = ZERO8;
    acc = wmma_bf16(a0, b0, acc);
    acc = wmma_bf16(a1, b1, acc);
    int mcol = hq * N_ + m0 + lr;
    float c_nk = nk[mcol], c_sk = sk[mcol];
#pragma unroll
    for (int j = 0; j < 8; j++) {
      float S = acc[j];
      float cosv = clampf(S / ((r_nq[j] + 1e-6f) * (c_nk + 1e-6f)), -0.95f, 0.95f);
      float covv = clampf((S - r_t[j] - r_mu[j] * (c_sk - chqv)) * COVSCALE, -50.f, 50.f);
      float d = ca * cosv + cb * covv + r_cc[j];
      dotsS[(j + 8 * lh) * 1024 + m0 + lr] = (__bf16)d;
    }
    b0 = nb0; b1 = nb1;
  }
  __syncthreads();

  // per-row softmax stats (wave w -> rows 2w, 2w+1)
  for (int rr = 0; rr < 2; rr++) {
    int r = wave * 2 + rr;
    float mx = -1e30f;
    for (int i = lane; i < 1024; i += 32) mx = fmaxf(mx, (float)dotsS[r * 1024 + i]);
    mx = redMax32(mx);
    float s = 0.f;
    for (int i = lane; i < 1024; i += 32) s += expf(((float)dotsS[r * 1024 + i] - mx) * invT);
    s = redAdd32(s);
    if (lane == 0) { rowMax[r] = mx; rowSI[r] = 1.0f / s; }
  }
  __syncthreads();

  // PV: out tile 16 x 64, K = 1024; wave = (colTile = w&3, K-half = w>>2)
  const int ct = wave & 3, half = wave >> 2;
  const int d0 = ct * 16;
  const __bf16* fvrow = fvT + ((size_t)hq * DH_ + d0 + lr) * N_;
  const float mx = rowMax[lr], rs = rowSI[lr];
  v8f acc = ZERO8;
  const int kbeg = half * 16;
  v16bf b = ldfrag(fvrow + kbeg * 32, lh);
  for (int kk = kbeg; kk < kbeg + 16; kk++) {
    int knext = kbeg + ((kk + 1 - kbeg) & 15);
    v16bf nb = ldfrag(fvrow + knext * 32, lh);
    v16bf a;
    int base = lr * 1024 + kk * 32 + 8 * lh;
#pragma unroll
    for (int i = 0; i < 8; i++) a[i]     = (__bf16)(expf(((float)dotsS[base + i] - mx) * invT) * rs);
#pragma unroll
    for (int i = 0; i < 8; i++) a[8 + i] = (__bf16)(expf(((float)dotsS[base + 16 + i] - mx) * invT) * rs);
    acc = wmma_bf16(a, b, acc);
    b = nb;
  }
  if (half == 1) {
#pragma unroll
    for (int j = 0; j < 8; j++) part[(ct * 32 + lane) * 8 + j] = acc[j];
  }
  __syncthreads();
  if (half == 0) {
#pragma unroll
    for (int j = 0; j < 8; j++) {
      float v = acc[j] + part[(ct * 32 + lane) * 8 + j];
      int r = j + 8 * lh;
      int col = h * DH_ + d0 + lr;
      fheadB[((size_t)qb * N_ + n0 + r) * (H_ * DH_) + col] = (__bf16)v;
    }
  }
}

// ---------------------------------------------------------------------------
// K9: output projection (bf16 WMMA) + bias -> f32 d_out
// ---------------------------------------------------------------------------
__global__ __launch_bounds__(256)
void outproj_kernel(const __bf16* __restrict__ fheadB, const __bf16* __restrict__ WoutT,
                    const float* __restrict__ b_out, float* __restrict__ out) {
  const int qb = blockIdx.y, n0 = blockIdx.x * 16;
  const int tid = threadIdx.x, wave = tid >> 5, lane = tid & 31;
  const int lr = lane & 15, lh = lane >> 4;
  const __bf16* arow = fheadB + ((size_t)qb * N_ + n0 + lr) * 512;
  __builtin_prefetch(WoutT + (size_t)(wave * 64 + lr) * 512, 0, 1);
  for (int ti = 0; ti < 4; ti++) {
    int ct = wave * 4 + ti;
    v8f acc = ZERO8;
    const __bf16* brow = WoutT + (size_t)(ct * 16 + lr) * 512;
    v16bf a = ldfrag(arow, lh);
    v16bf b = ldfrag(brow, lh);
    for (int kk = 0; kk < 16; kk++) {
      int kn = ((kk + 1) & 15) * 32;
      v16bf na = ldfrag(arow + kn, lh);
      v16bf nb = ldfrag(brow + kn, lh);
      acc = wmma_bf16(a, b, acc);
      a = na; b = nb;
    }
    int col = ct * 16 + lr;
    float bb = b_out[col];
#pragma unroll
    for (int j = 0; j < 8; j++)
      out[((size_t)qb * N_ + n0 + j + 8 * lh) * DIM_ + col] = acc[j] + bb;
  }
}

// ---------------------------------------------------------------------------
extern "C" void kernel_launch(void* const* d_in, const int* in_sizes, int n_in,
                              void* d_out, int out_size, void* d_ws, size_t ws_size,
                              hipStream_t stream) {
  (void)in_sizes; (void)n_in; (void)out_size; (void)ws_size;
  const float* q      = (const float*)d_in[0];
  const float* k      = (const float*)d_in[1];
  const float* v      = (const float*)d_in[2];
  const float* ln_w   = (const float*)d_in[3];
  const float* ln_b   = (const float*)d_in[4];
  const float* W_in   = (const float*)d_in[5];
  const float* W_out  = (const float*)d_in[6];
  const float* b_out  = (const float*)d_in[7];
  const float* wp_W1  = (const float*)d_in[8];
  const float* wp_b1  = (const float*)d_in[9];
  const float* wp_lnw = (const float*)d_in[10];
  const float* wp_lnb = (const float*)d_in[11];
  const float* wp_W2  = (const float*)d_in[12];
  const float* wp_b2  = (const float*)d_in[13];
  const float* wp_W3  = (const float*)d_in[14];
  const float* wp_b3  = (const float*)d_in[15];
  const float* wp_W4  = (const float*)d_in[16];
  const float* wp_b4  = (const float*)d_in[17];
  const float* wtemp  = (const float*)d_in[18];
  float* out = (float*)d_out;

  char* base = (char*)d_ws;
  size_t off = 0;
  auto take = [&](size_t bytes) -> void* {
    void* p = base + off;
    off += (bytes + 255) & ~(size_t)255;
    return p;
  };
  const size_t FEL = (size_t)HQ_ * N_ * DH_;          // 2M elements
  __bf16* fqB    = (__bf16*)take(FEL * 2);
  __bf16* fkB    = (__bf16*)take(FEL * 2);
  __bf16* fvT    = (__bf16*)take(FEL * 2);
  __bf16* fheadB = (__bf16*)take((size_t)TOK_ * 512 * 2);
  __bf16* WinT   = (__bf16*)take((size_t)512 * 512 * 2);
  __bf16* WoutT  = (__bf16*)take((size_t)512 * 512 * 2);
  float*  nq     = (float*)take((size_t)HQ_ * N_ * 4);
  float*  sq     = (float*)take((size_t)HQ_ * N_ * 4);
  float*  nk     = (float*)take((size_t)HQ_ * N_ * 4);
  float*  sk     = (float*)take((size_t)HQ_ * N_ * 4);
  float*  tdot   = (float*)take((size_t)HQ_ * N_ * 4);
  float*  vrow   = (float*)take((size_t)HQ_ * N_ * 4);
  float*  fkmean = (float*)take((size_t)HQ_ * DH_ * 4);
  float*  chq    = (float*)take((size_t)HQ_ * 4);
  float*  feat   = (float*)take((size_t)H_ * 128 * 4);
  float*  wht    = (float*)take((size_t)H_ * 3 * 4);
  float*  coefs  = (float*)take(24 * 4);
  double* accum  = (double*)take(6 * 8);
  float*  invT   = (float*)take(4);

  zero_accum_kernel<<<1, 32, 0, stream>>>(accum);
  convw_kernel<<<2048, 256, 0, stream>>>(W_in, W_out, WinT, WoutT);

  dim3 gproj(64, 4);
  proj_kernel<<<gproj, 256, 0, stream>>>(q, ln_w, ln_b, WinT, fqB, nq, sq, 0);
  proj_kernel<<<gproj, 256, 0, stream>>>(k, ln_w, ln_b, WinT, fkB, nk, sk, 1);
  proj_kernel<<<gproj, 256, 0, stream>>>(v, ln_w, ln_b, WinT, fvT, nullptr, nullptr, 2);

  colmean_kernel<<<8, 256, 0, stream>>>(fkB, fkmean);
  chq_kernel<<<1, 32, 0, stream>>>(fkmean, chq);
  feat_kernel<<<4, 256, 0, stream>>>(fqB, fkB, feat);
  tdot_kernel<<<128, 256, 0, stream>>>(fqB, fkmean, tdot);
  mlp_kernel<<<1, 256, 0, stream>>>(feat, wp_W1, wp_b1, wp_lnw, wp_lnb,
                                    wp_W2, wp_b2, wp_W3, wp_b3, wp_W4, wp_b4, wtemp, wht);

  dim3 gsc(64, 4, 8);
  score_pass_kernel<<<gsc, 256, 0, stream>>>(fqB, fkB, nq, sq, nk, sk, tdot, chq,
                                             vrow, coefs, accum, 0);
  finalize1_kernel<<<1, 256, 0, stream>>>(vrow, accum, wht, coefs);
  score_pass_kernel<<<gsc, 256, 0, stream>>>(fqB, fkB, nq, sq, nk, sk, tdot, chq,
                                             vrow, coefs, accum, 1);
  finalize2_kernel<<<1, 32, 0, stream>>>(accum, invT);

  softmax_pv_kernel<<<gsc, 256, 0, stream>>>(fqB, fkB, fvT, nq, sq, nk, sk, tdot, chq,
                                             vrow, coefs, invT, fheadB);
  outproj_kernel<<<gproj, 256, 0, stream>>>(fheadB, WoutT, b_out, out);
}
